// AutoReg_2293512536244
// MI455X (gfx1250) — compile-verified, matrix operand order checked by reading, not run on hardware
//
#include <hip/hip_runtime.h>
#include <hip/hip_bf16.h>

// ---------------------------------------------------------------------------
// CDNA5 / gfx1250 transformer forward (bf16 WMMA path)
//   wave32, v_wmma_f32_16x16x32_bf16, double-buffered LDS GEMM with async
//   global->LDS staging (ASYNCcnt), flash attention
// ---------------------------------------------------------------------------

typedef __attribute__((ext_vector_type(16))) __bf16 v16bf;
typedef __attribute__((ext_vector_type(8)))  __bf16 v8bf;
typedef __attribute__((ext_vector_type(8)))  float  v8f;
typedef __attribute__((ext_vector_type(4)))  int    v4i_;

#ifndef __has_builtin
#define __has_builtin(x) 0
#endif
#if __has_builtin(__builtin_amdgcn_global_load_async_to_lds_b128)
#define HAVE_ASYNC_LDS 1
#else
#define HAVE_ASYNC_LDS 0
#endif

#if HAVE_ASYNC_LDS
// clang signature (from probe): (int4 AS1*, int4 AS3*, imm offset, imm cpol)
typedef v4i_ __attribute__((address_space(1)))* gv4_ptr;
typedef v4i_ __attribute__((address_space(3)))* lv4_ptr;
__device__ __forceinline__ void async_g2l_b128(const void* g, void* l) {
  __builtin_amdgcn_global_load_async_to_lds_b128((gv4_ptr)g, (lv4_ptr)l, 0, 0);
}
#endif
__device__ __forceinline__ void wait_async() {
#if __has_builtin(__builtin_amdgcn_s_wait_asynccnt)
  __builtin_amdgcn_s_wait_asynccnt(0);
#elif HAVE_ASYNC_LDS
  asm volatile("s_wait_asynccnt 0x0" ::: "memory");
#endif
}

#define LNUM 4
#define BNUM 4
#define SLEN 1024
#define DMOD 1024
#define HNUM 16
#define FMLP 4096
#define HDIM 64
#define NOUT 50
#define MROW (BNUM * SLEN)   // 4096

// epilogue flags (compile-time template parameter)
#define GF_RELU   1
#define GF_RESID  2   // fp32 residual accumulate into Hres
#define GF_VTRANS 4   // bf16 store transposed to [B,H,HD,S]
#define GF_BF16   8   // bf16 store row-major [M,N]
#define GF_BIAS   16

__device__ __forceinline__ __bf16 f2bf(float f) {
  unsigned int u = __builtin_bit_cast(unsigned int, f);
  u += 0x7FFFu + ((u >> 16) & 1u);           // round-to-nearest-even
  return __builtin_bit_cast(__bf16, (unsigned short)(u >> 16));
}
__device__ __forceinline__ float bf2f(__bf16 b) {
  unsigned int u = ((unsigned int)__builtin_bit_cast(unsigned short, b)) << 16;
  return __builtin_bit_cast(float, u);
}
// ISA 7.12.2 16-bit A/B fragment: per lane, vector elems 0..7 <-> K = koff..koff+7,
// elems 8..15 <-> K = koff+16..koff+23, with koff = 8*(lane>=16). Two b128 loads.
__device__ __forceinline__ v16bf load_frag(const __bf16* p0, const __bf16* p1) {
  v8bf lo = *(const v8bf*)p0;
  v8bf hi = *(const v8bf*)p1;
  v16bf r;
#pragma unroll
  for (int i = 0; i < 8; ++i) { r[i] = lo[i]; r[i + 8] = hi[i]; }
  return r;
}
__device__ __forceinline__ v8f wmma_bf16(v16bf a, v16bf b, v8f c) {
  return __builtin_amdgcn_wmma_f32_16x16x32_bf16(false, a, false, b, (short)0, c,
                                                 false, false);
}
__device__ __forceinline__ v8f v8f_zero() {
  v8f z;
#pragma unroll
  for (int i = 0; i < 8; ++i) z[i] = 0.0f;
  return z;
}

// --------------------------- weight fp32 -> bf16 ---------------------------
__global__ __launch_bounds__(256) void cvt_kernel(const float* __restrict__ in,
                                                  __bf16* __restrict__ out, int n) {
  int i = blockIdx.x * 256 + threadIdx.x;
  if (i < n) out[i] = f2bf(in[i]);
}

// --------------------------- encoder / embeddings --------------------------
__global__ __launch_bounds__(256) void encode_kernel(
    const float* __restrict__ emb1, const float* __restrict__ emb2,
    const float* __restrict__ wn1, const float* __restrict__ wn2,
    const float* __restrict__ pos, const float* __restrict__ bos,
    const float* __restrict__ n1, const float* __restrict__ n2,
    const int* __restrict__ c1, const int* __restrict__ c2,
    float* __restrict__ h) {
  int idx = blockIdx.x * 256 + threadIdx.x;            // < B*S*D
  int d = idx & (DMOD - 1);
  int s = (idx >> 10) & (SLEN - 1);
  int b = idx >> 20;
  float v;
  if (s == 0) {
    v = bos[d];                                        // teacher-forcing BOS
  } else {
    int t = s - 1;
    int i1 = c1[b * SLEN + t], i2 = c2[b * SLEN + t];
    v = emb1[i1 * DMOD + d] + emb2[i2 * DMOD + d]
      + n1[b * SLEN + t] * wn1[d] + n2[b * SLEN + t] * wn2[d]
      + pos[t * DMOD + d];
  }
  h[idx] = v;
}

// --------------------------------- LayerNorm -------------------------------
__global__ __launch_bounds__(256) void ln_kernel(const float* __restrict__ X,
                                                 const float* __restrict__ g,
                                                 const float* __restrict__ bt,
                                                 __bf16* __restrict__ O) {
  __shared__ float sa[256], sb[256];
  const int tid = threadIdx.x;
  const float* row = X + (size_t)blockIdx.x * DMOD;
  float vals[4], lsum = 0.f, lsq = 0.f;
#pragma unroll
  for (int i = 0; i < 4; ++i) {
    float v = row[tid + i * 256];
    vals[i] = v; lsum += v; lsq += v * v;
  }
  sa[tid] = lsum; sb[tid] = lsq;
  __syncthreads();
  for (int s = 128; s > 0; s >>= 1) {
    if (tid < s) { sa[tid] += sa[tid + s]; sb[tid] += sb[tid + s]; }
    __syncthreads();
  }
  float mean = sa[0] * (1.0f / DMOD);
  float var  = sb[0] * (1.0f / DMOD) - mean * mean;
  float rstd = rsqrtf(var + 1e-6f);
  __bf16* orow = O + (size_t)blockIdx.x * DMOD;
#pragma unroll
  for (int i = 0; i < 4; ++i) {
    int d = tid + i * 256;
    orow[d] = f2bf((vals[i] - mean) * rstd * g[d] + bt[d]);
  }
}

// ------------------------------ WMMA GEMM ----------------------------------
// C[M,N] = alpha * A[M,K]@W[K,N] (+bias)(+relu), fused compile-time outputs.
// Block: 256 threads (8 waves). Tile: 128x128, K-step 32, double-buffered LDS.
// A tile staged via GLOBAL_LOAD_ASYNC_TO_LDS_B128 (ASYNCcnt) when available;
// W tile register-staged with a transpose scatter so B-fragments are b128.
// Wave (rowg 0..3, colg 0..1): rows [32*rowg, +32) x cols [64*colg, +64)
//   -> 2 A-frags x 4 B-frags = 8 wmma / K-step, each B-frag reused twice.
template <int FLAGS>
__global__ __launch_bounds__(256) void gemm_kernel(
    const __bf16* __restrict__ A, const __bf16* __restrict__ W,
    const float* __restrict__ bias, float* __restrict__ Hres,
    __bf16* __restrict__ Obf, int M, int N, int K, float alpha) {
  __shared__ __bf16 Al[2][128 * 32];   // A tile, row-major [128][32]
  __shared__ __bf16 Wl[2][128 * 32];   // W tile transposed [128(n)][32(k)]
  const int tid  = threadIdx.x;
  const int wave = tid >> 5;
  const int lane = tid & 31;
  const int half = lane >> 4;
  const int lr   = lane & 15;
  const int koff = half * 8;
  const int rowg = wave >> 1;          // 0..3
  const int colg = wave & 1;           // 0..1
  const int m0 = blockIdx.y * 128;
  const int n0 = blockIdx.x * 128;

  // staging assignments (each thread: 16 A elems + 16 W elems per K-step)
  const int arow = tid >> 1, acol = (tid & 1) * 16;
  const int wkk = tid >> 3;            // 0..31 (k within tile)
  const int wnn = (tid & 7) * 16;      // 0..112 (n within tile)

  v8f acc[2][4];
#pragma unroll
  for (int s = 0; s < 2; ++s)
#pragma unroll
    for (int c = 0; c < 4; ++c) acc[s][c] = v8f_zero();

  v8bf ar0, ar1, wr0, wr1;
  // A tile: issue copy (async: straight to LDS; fallback: into regs)
  auto a_issue = [&](int k0, int buf) {
    const __bf16* asrc = A + (size_t)(m0 + arow) * K + k0 + acol;
#if HAVE_ASYNC_LDS
    async_g2l_b128(asrc,     &Al[buf][arow * 32 + acol]);
    async_g2l_b128(asrc + 8, &Al[buf][arow * 32 + acol + 8]);
#else
    ar0 = *(const v8bf*)asrc;
    ar1 = *(const v8bf*)(asrc + 8);
    (void)buf;
#endif
  };
  auto a_commit = [&](int buf) {
#if !HAVE_ASYNC_LDS
    *(v8bf*)&Al[buf][arow * 32 + acol]     = ar0;
    *(v8bf*)&Al[buf][arow * 32 + acol + 8] = ar1;
#else
    (void)buf;
#endif
  };
  auto w_gload = [&](int k0) {
    const __bf16* wsrc = W + (size_t)(k0 + wkk) * N + n0 + wnn;
    wr0 = *(const v8bf*)wsrc;
    wr1 = *(const v8bf*)(wsrc + 8);
  };
  auto w_lstore = [&](int buf) {
#pragma unroll
    for (int i = 0; i < 8; ++i) Wl[buf][(wnn + i) * 32 + wkk] = wr0[i];
#pragma unroll
    for (int i = 0; i < 8; ++i) Wl[buf][(wnn + 8 + i) * 32 + wkk] = wr1[i];
  };

  a_issue(0, 0);
  w_gload(0);
  a_commit(0);
  w_lstore(0);
  wait_async();
  __syncthreads();
  const int nsteps = K >> 5;
  for (int step = 0; step < nsteps; ++step) {
    const int cur = step & 1;
    const bool has_next = (step + 1) < nsteps;
    if (has_next) {                         // overlap with wmma below
      a_issue((step + 1) << 5, cur ^ 1);
      w_gload((step + 1) << 5);
    }
    const int r0 = rowg * 32 + lr;
    v16bf af0 = load_frag(&Al[cur][r0 * 32 + koff],
                          &Al[cur][r0 * 32 + koff + 16]);
    v16bf af1 = load_frag(&Al[cur][(r0 + 16) * 32 + koff],
                          &Al[cur][(r0 + 16) * 32 + koff + 16]);
#pragma unroll
    for (int c = 0; c < 4; ++c) {
      const int n = colg * 64 + c * 16 + lr;
      v16bf bf = load_frag(&Wl[cur][n * 32 + koff],
                           &Wl[cur][n * 32 + koff + 16]);
      acc[0][c] = wmma_bf16(af0, bf, acc[0][c]);
      acc[1][c] = wmma_bf16(af1, bf, acc[1][c]);
    }
    if (has_next) {                         // other buffer: no read conflict
      a_commit(cur ^ 1);
      w_lstore(cur ^ 1);
    }
    wait_async();                           // own copies done before barrier
    __syncthreads();
  }
  // epilogue (straight-line; FLAGS is compile-time)
#pragma unroll
  for (int sub = 0; sub < 2; ++sub) {
#pragma unroll
    for (int c = 0; c < 4; ++c) {
#pragma unroll
      for (int r = 0; r < 8; ++r) {
        int m = m0 + rowg * 32 + sub * 16 + half * 8 + r;
        int n = n0 + colg * 64 + c * 16 + lr;
        float v = acc[sub][c][r] * alpha;
        if constexpr (FLAGS & GF_BIAS) v += bias[n];
        if constexpr (FLAGS & GF_RELU) v = fmaxf(v, 0.f);
        if constexpr (FLAGS & GF_RESID) {
          size_t ix = (size_t)m * N + n;
          Hres[ix] = Hres[ix] + v;
        }
        if constexpr (FLAGS & GF_BF16) Obf[(size_t)m * N + n] = f2bf(v);
        if constexpr (FLAGS & GF_VTRANS) {  // V stored [B,H,HD,S]
          int b = m >> 10, s = m & (SLEN - 1);
          int hh = n >> 6, hd = n & (HDIM - 1);
          Obf[((size_t)((b * HNUM + hh) * HDIM + hd) << 10) + s] = f2bf(v);
        }
      }
    }
  }
}

// --------------------------- flash attention -------------------------------
// 1 wave / block; block = (b, h, 16-query tile). Q bf16 [B,S,D] (pre-scaled),
// K bf16 [B,S,D], V bf16 transposed [B,H,HD,S]. Out bf16 [B,S,D].
__global__ __launch_bounds__(32) void attn_kernel(
    const __bf16* __restrict__ Q, const __bf16* __restrict__ Km,
    const __bf16* __restrict__ Vt, __bf16* __restrict__ O) {
  __shared__ __bf16 Pl[16 * 32];
  const int bid = blockIdx.x;
  const int qt = bid & 63;
  const int h  = (bid >> 6) & (HNUM - 1);
  const int b  = bid >> 10;
  const int lane = threadIdx.x;
  const int half = lane >> 4, lr = lane & 15;
  const int koff = half * 8;
  const int qbase = qt * 16;

  const __bf16* qrow = Q + ((size_t)(b * SLEN + qbase + lr) << 10) + h * HDIM;
  v16bf qa0 = load_frag(qrow + koff,      qrow + koff + 16);
  v16bf qa1 = load_frag(qrow + 32 + koff, qrow + 32 + koff + 16);

  v8f o[4];
  float rmax[8], rsum[8];
#pragma unroll
  for (int c = 0; c < 4; ++c) o[c] = v8f_zero();
#pragma unroll
  for (int r = 0; r < 8; ++r) { rmax[r] = -3.0e38f; rsum[r] = 0.f; }

  for (int j0 = 0; j0 * 16 <= qbase + 15; j0 += 2) {
    // scores chunk 16x32 = two C tiles (4 wmma)
    v8f s[2];
#pragma unroll
    for (int t = 0; t < 2; ++t) {
      const __bf16* krow =
          Km + ((size_t)(b * SLEN + (j0 + t) * 16 + lr) << 10) + h * HDIM;
      v16bf kb0 = load_frag(krow + koff,      krow + koff + 16);
      v16bf kb1 = load_frag(krow + 32 + koff, krow + 32 + koff + 16);
      s[t] = v8f_zero();
      s[t] = wmma_bf16(qa0, kb0, s[t]);
      s[t] = wmma_bf16(qa1, kb1, s[t]);
    }
    __syncthreads();   // Pl reads of previous iter complete before overwrite
#pragma unroll
    for (int r = 0; r < 8; ++r) {
      int qg = qbase + half * 8 + r;
      int kg0 = j0 * 16 + lr, kg1 = kg0 + 16;
      float v0 = (kg0 <= qg) ? s[0][r] : -1e30f;   // causal mask
      float v1 = (kg1 <= qg) ? s[1][r] : -1e30f;
      float m = v0 > v1 ? v0 : v1;
#pragma unroll
      for (int d = 1; d < 16; d <<= 1) {           // row max over 16 lanes
        float t = __shfl_xor(m, d, 32);
        m = m > t ? m : t;
      }
      float newm = rmax[r] > m ? rmax[r] : m;
      float corr = __expf(rmax[r] - newm);
      float p0 = __expf(v0 - newm);
      float p1 = __expf(v1 - newm);
      float ls = p0 + p1;
#pragma unroll
      for (int d = 1; d < 16; d <<= 1) ls += __shfl_xor(ls, d, 32);
      rsum[r] = rsum[r] * corr + ls;
      rmax[r] = newm;
#pragma unroll
      for (int c = 0; c < 4; ++c) o[c][r] = o[c][r] * corr;
      int rm = half * 8 + r;
      Pl[rm * 32 + lr]      = f2bf(p0);
      Pl[rm * 32 + 16 + lr] = f2bf(p1);
    }
    __syncthreads();
    // P(16x32) @ V(32x64): A-frag from LDS, contiguous Vt B-frags (4 wmma)
    v16bf pa = load_frag(&Pl[lr * 32 + koff], &Pl[lr * 32 + koff + 16]);
#pragma unroll
    for (int c = 0; c < 4; ++c) {
      int hd = c * 16 + lr;
      const __bf16* vrow =
          Vt + ((size_t)((b * HNUM + h) * HDIM + hd) << 10) + j0 * 16;
      v16bf vb = load_frag(vrow + koff, vrow + 16 + koff);
      o[c] = wmma_bf16(pa, vb, o[c]);
    }
  }
#pragma unroll
  for (int c = 0; c < 4; ++c) {
#pragma unroll
    for (int r = 0; r < 8; ++r) {
      int m = b * SLEN + qbase + half * 8 + r;
      int n = h * HDIM + c * 16 + lr;
      O[((size_t)m << 10) + n] = f2bf(o[c][r] / rsum[r]);
    }
  }
}

// ------------------------------ decoder head -------------------------------
__global__ __launch_bounds__(256) void head_kernel(const __bf16* __restrict__ Xf,
                                                   const float* __restrict__ wd,
                                                   const float* __restrict__ bd,
                                                   float* __restrict__ out) {
  int idx = blockIdx.x * 256 + threadIdx.x;
  if (idx >= MROW * NOUT) return;
  int o = idx % NOUT, m = idx / NOUT;
  const __bf16* x = Xf + ((size_t)m << 10);
  float acc = bd[o];
  for (int k = 0; k < DMOD; ++k) acc += bf2f(x[k]) * wd[k * NOUT + o];
  out[idx] = acc;
}

// ------------------------------ host launch --------------------------------
extern "C" void kernel_launch(void* const* d_in, const int* in_sizes, int n_in,
                              void* d_out, int out_size, void* d_ws, size_t ws_size,
                              hipStream_t stream) {
  const float* emb1 = (const float*)d_in[0];
  const float* emb2 = (const float*)d_in[1];
  const float* wn1  = (const float*)d_in[2];
  const float* wn2  = (const float*)d_in[3];
  const float* pos  = (const float*)d_in[4];
  const float* bos  = (const float*)d_in[5];
  const float* ln1g = (const float*)d_in[6];
  const float* ln1b = (const float*)d_in[7];
  const float* ln2g = (const float*)d_in[8];
  const float* ln2b = (const float*)d_in[9];
  const float* lnfg = (const float*)d_in[10];
  const float* lnfb = (const float*)d_in[11];
  const float* wq = (const float*)d_in[12];
  const float* wk = (const float*)d_in[13];
  const float* wv = (const float*)d_in[14];
  const float* wo = (const float*)d_in[15];
  const float* w1 = (const float*)d_in[16];
  const float* b1 = (const float*)d_in[17];
  const float* w2 = (const float*)d_in[18];
  const float* b2 = (const float*)d_in[19];
  const float* wd = (const float*)d_in[20];
  const float* bd = (const float*)d_in[21];
  const float* tn1 = (const float*)d_in[22];
  const float* tn2 = (const float*)d_in[23];
  const int*   tc1 = (const int*)d_in[24];
  const int*   tc2 = (const int*)d_in[25];

  // workspace carve-up (256B aligned); total ~201 MB
  char* p = (char*)d_ws;
  auto take = [&](size_t bytes) -> char* {
    char* r = p;
    p += (bytes + 255) & ~(size_t)255;
    return r;
  };
  const size_t LDD = (size_t)LNUM * DMOD * DMOD;
  const size_t LDF = (size_t)LNUM * DMOD * FMLP;
  __bf16* wqb = (__bf16*)take(LDD * 2);
  __bf16* wkb = (__bf16*)take(LDD * 2);
  __bf16* wvb = (__bf16*)take(LDD * 2);
  __bf16* wob = (__bf16*)take(LDD * 2);
  __bf16* w1b = (__bf16*)take(LDF * 2);
  __bf16* w2b = (__bf16*)take(LDF * 2);
  float*  h   = (float*)take((size_t)MROW * DMOD * 4);
  __bf16* xb  = (__bf16*)take((size_t)MROW * DMOD * 2);
  __bf16* qb  = (__bf16*)take((size_t)MROW * DMOD * 2);
  __bf16* kb  = (__bf16*)take((size_t)MROW * DMOD * 2);
  __bf16* vtb = (__bf16*)take((size_t)MROW * DMOD * 2);
  __bf16* aob = (__bf16*)take((size_t)MROW * DMOD * 2);
  __bf16* actb= (__bf16*)take((size_t)MROW * FMLP * 2);
  __bf16* xfb = (__bf16*)take((size_t)MROW * DMOD * 2);

  auto cvt = [&](const float* src, __bf16* dst, size_t n) {
    cvt_kernel<<<(unsigned)((n + 255) / 256), 256, 0, stream>>>(src, dst, (int)n);
  };
  cvt(wq, wqb, LDD); cvt(wk, wkb, LDD); cvt(wv, wvb, LDD); cvt(wo, wob, LDD);
  cvt(w1, w1b, LDF); cvt(w2, w2b, LDF);

  encode_kernel<<<(MROW * DMOD) / 256, 256, 0, stream>>>(
      emb1, emb2, wn1, wn2, pos, bos, tn1, tn2, tc1, tc2, h);

  const dim3 gD(DMOD / 128, MROW / 128);   // (8,32)
  const dim3 gF(FMLP / 128, MROW / 128);   // (32,32)
  const float qscale = 0.125f;             // 1/sqrt(HD)

  for (int l = 0; l < LNUM; ++l) {
    const size_t oDD = (size_t)l * DMOD * DMOD;
    const size_t oDF = (size_t)l * DMOD * FMLP;
    ln_kernel<<<MROW, 256, 0, stream>>>(h, ln1g + l * DMOD, ln1b + l * DMOD, xb);
    gemm_kernel<GF_BF16><<<gD, 256, 0, stream>>>(
        xb, wqb + oDD, nullptr, nullptr, qb, MROW, DMOD, DMOD, qscale);
    gemm_kernel<GF_BF16><<<gD, 256, 0, stream>>>(
        xb, wkb + oDD, nullptr, nullptr, kb, MROW, DMOD, DMOD, 1.0f);
    gemm_kernel<GF_VTRANS><<<gD, 256, 0, stream>>>(
        xb, wvb + oDD, nullptr, nullptr, vtb, MROW, DMOD, DMOD, 1.0f);
    attn_kernel<<<BNUM * HNUM * (SLEN / 16), 32, 0, stream>>>(qb, kb, vtb, aob);
    gemm_kernel<GF_RESID><<<gD, 256, 0, stream>>>(
        aob, wob + oDD, nullptr, h, nullptr, MROW, DMOD, DMOD, 1.0f);
    ln_kernel<<<MROW, 256, 0, stream>>>(h, ln2g + l * DMOD, ln2b + l * DMOD, xb);
    gemm_kernel<GF_BF16 | GF_BIAS | GF_RELU><<<gF, 256, 0, stream>>>(
        xb, w1b + oDF, b1 + l * FMLP, nullptr, actb, MROW, FMLP, DMOD, 1.0f);
    gemm_kernel<GF_RESID | GF_BIAS><<<gD, 256, 0, stream>>>(
        actb, w2b + oDF, b2 + l * DMOD, h, nullptr, MROW, DMOD, FMLP, 1.0f);
  }
  ln_kernel<<<MROW, 256, 0, stream>>>(h, lnfg, lnfb, xfb);
  head_kernel<<<(MROW * NOUT + 255) / 256, 256, 0, stream>>>(
      xfb, wd, bd, (float*)d_out);
}
